// ParticleGNNImprovedLite_25812753449157
// MI455X (gfx1250) — matrix operand order repaired and checked by reading, not compiled, not run on hardware
//
#include <hip/hip_runtime.h>

// ---------------------------------------------------------------------------
// GAT network forward for gfx1250 (MI455X). Dense projections use
// v_wmma_f32_16x16x32_f16 with K/OUT/bias/relu as template params so the
// whole kernel (loads, epilogue stores) is branchless constant-offset code;
// edge softmax/aggregation uses atomics (memory bound: ~3-4 GB of traffic,
// largely resident in the 192MB L2).
// ---------------------------------------------------------------------------

typedef __attribute__((ext_vector_type(16))) _Float16 v16h;
typedef __attribute__((ext_vector_type(8)))  float    v8f;
typedef __attribute__((ext_vector_type(4)))  float    vf4;

#define HEADS 4
#define EPS 1e-5f
#define NEG_SLOPE 0.2f

// ---------------------------------------------------------------------------
// WMMA GEMM: Y[M,OUT] = X[M,K] @ W[OUT,K]^T (+bias) (optional relu)
// One wave per 16x16 output tile; K in {16,32,64}, OUT in {32,64,128,256}.
// A layout (16x32 f16): lane half=lane>>4, m=lane&15; elems 0..7 -> K=k0+8h+j,
// elems 8..15 -> K=k0+16+8h+j.  B layout (32x16): col n=lane&15, elems 0..15
// -> K same as A (k0+8h+j, k0+16+8h+j).  D layout: vgpr v -> row v+8h, col lane&15.
// No divergent branches in the hot path: A row pointer is clamped (bad rows
// are discarded by the store guard in the remainder path), B out-of-range
// half selects zero via cndmask, full tiles store unguarded.
// ---------------------------------------------------------------------------
template<int K, int OUT, bool HAS_BIAS, bool RELU>
__global__ __launch_bounds__(32)
void wmma_gemm_t(const float* __restrict__ X, const float* __restrict__ W,
                 const float* __restrict__ bias, float* __restrict__ Y, int M)
{
    const int lane = threadIdx.x & 31;
    const int half = lane >> 4;
    const int m16  = lane & 15;
    const int rowBase = blockIdx.x * 16;
    const int colBase = blockIdx.y * 16;
    const int row  = rowBase + m16;
    const int rowC = row < M ? row : (M - 1);   // clamp; no branch
    const int col  = colBase + m16;             // OUT is a multiple of 16

    v8f acc = {};
#pragma unroll
    for (int k0 = 0; k0 < K; k0 += 32) {
        // ---- A fragment: 4x vf4, unconditional vector loads ----
        const float* xp = X + (size_t)rowC * K + k0 + half * 8;
        vf4 a0 = *(const vf4*)(xp);
        vf4 a1 = *(const vf4*)(xp + 4);
        vf4 a2, a3;
        if (k0 + 32 <= K) {                      // compile-time after unroll
            a2 = *(const vf4*)(xp + 16);
            a3 = *(const vf4*)(xp + 20);
        } else {
            a2 = (vf4)0.0f; a3 = (vf4)0.0f;
        }
        // ---- B fragment: 4x vf4, same K order as A (8h+j, 16+8h+j) ---
        const float* wp = W + (size_t)col * K + k0 + half * 8;
        vf4 b0 = *(const vf4*)(wp);
        vf4 b1 = *(const vf4*)(wp + 4);
        vf4 b2, b3;
        if (k0 + 32 <= K) {
            b2 = *(const vf4*)(wp + 16);
            b3 = *(const vf4*)(wp + 20);
        } else {
            b2 = (vf4)0.0f; b3 = (vf4)0.0f;
        }

        v16h a, b;
#pragma unroll
        for (int j = 0; j < 4; ++j) {
            a[j]      = (_Float16)a0[j];
            a[4 + j]  = (_Float16)a1[j];
            a[8 + j]  = (_Float16)a2[j];
            a[12 + j] = (_Float16)a3[j];
            b[j]      = (_Float16)b0[j];
            b[4 + j]  = (_Float16)b1[j];
            b[8 + j]  = (_Float16)b2[j];
            b[12 + j] = (_Float16)b3[j];
        }
        acc = __builtin_amdgcn_wmma_f32_16x16x32_f16(
            /*neg_a=*/false, a, /*neg_b=*/false, b,
            /*c_mod=*/(short)0, acc, /*reuse_a=*/false, /*reuse_b=*/false);
    }

    // ---- epilogue: bias loaded once; full tiles store unguarded ----
    const float bv = HAS_BIAS ? bias[col] : 0.0f;
    const int r0 = rowBase + half * 8;
    if (rowBase + 16 <= M) {                     // block-uniform branch
        float* yp = Y + (size_t)r0 * OUT + col;
#pragma unroll
        for (int v = 0; v < 8; ++v) {
            float y = acc[v] + bv;
            if (RELU) y = fmaxf(y, 0.0f);
            yp[(size_t)v * OUT] = y;             // constant immediate offsets
        }
    } else {
#pragma unroll
        for (int v = 0; v < 8; ++v) {
            int r = r0 + v;
            if (r < M) {
                float y = acc[v] + bv;
                if (RELU) y = fmaxf(y, 0.0f);
                Y[(size_t)r * OUT + col] = y;
            }
        }
    }
}

template<int K, int OUT, bool HAS_BIAS, bool RELU>
static inline void launch_gemm(const float* X, const float* W, const float* bias, float* Y,
                               int M, hipStream_t stream)
{
    dim3 g((unsigned)((M + 15) / 16), (unsigned)(OUT / 16));
    wmma_gemm_t<K, OUT, HAS_BIAS, RELU><<<g, 32, 0, stream>>>(X, W, bias, Y, M);
}

// ---------------------------------------------------------------------------
// Utility fills / reductions
// ---------------------------------------------------------------------------
__global__ void fill_f32_kernel(float* p, float v, long long n) {
    long long i = (long long)blockIdx.x * blockDim.x + threadIdx.x;
    if (i < n) p[i] = v;
}
__global__ void fill_i32_kernel(int* p, int v, long long n) {
    long long i = (long long)blockIdx.x * blockDim.x + threadIdx.x;
    if (i < n) p[i] = v;
}

__global__ void reduce_sum_kernel(const float* __restrict__ x, long long n, float* out) {
    __shared__ float sd[256];
    long long i = (long long)blockIdx.x * blockDim.x + threadIdx.x;
    long long stride = (long long)gridDim.x * blockDim.x;
    float s = 0.0f;
    for (; i < n; i += stride) s += x[i];
    sd[threadIdx.x] = s;
    __syncthreads();
    for (int o = 128; o > 0; o >>= 1) {
        if ((int)threadIdx.x < o) sd[threadIdx.x] += sd[threadIdx.x + o];
        __syncthreads();
    }
    if (threadIdx.x == 0) atomicAdd(out, sd[0]);
}

// scal[0] = sum(edge_attr) (from reduce). This kernel fills:
// scal[1..4]  = k_edge head scalars for conv1 (C=32)
// scal[5..8]  = conv2 (C=64), scal[9..12] = conv3 (C=32)
// scal[13]    = mean(edge_attr)
__global__ void finalize_scalars_kernel(const float* le1, const float* ae1,
                                        const float* le2, const float* ae2,
                                        const float* le3, const float* ae3,
                                        float* scal, float invE)
{
    int t = threadIdx.x;
    if (t < 4) {
        float s = 0.0f;
        for (int c = 0; c < 32; ++c) s += le1[t * 32 + c] * ae1[t * 32 + c];
        scal[1 + t] = s;
    } else if (t < 8) {
        int h = t - 4; float s = 0.0f;
        for (int c = 0; c < 64; ++c) s += le2[h * 64 + c] * ae2[h * 64 + c];
        scal[5 + h] = s;
    } else if (t < 12) {
        int h = t - 8; float s = 0.0f;
        for (int c = 0; c < 32; ++c) s += le3[h * 32 + c] * ae3[h * 32 + c];
        scal[9 + h] = s;
    } else if (t == 12) {
        scal[13] = scal[0] * invE;
    }
}

// ---------------------------------------------------------------------------
// Per-node attention coefficients: a_src/a_dst [N,H]
// ---------------------------------------------------------------------------
__global__ void attn_node_kernel(const float* __restrict__ P,
                                 const float* __restrict__ att_src,
                                 const float* __restrict__ att_dst,
                                 float* __restrict__ a_src, float* __restrict__ a_dst,
                                 int N, int C)
{
    long long i = (long long)blockIdx.x * blockDim.x + threadIdx.x;
    if (i >= (long long)N * HEADS) return;
    int h = (int)(i & 3);
    long long n = i >> 2;
    const float* row = P + n * (long long)(HEADS * C) + (long long)h * C;
    float ss = 0.0f, sd = 0.0f;
    for (int c = 0; c < C; ++c) {
        float v = row[c];
        ss += v * att_src[h * C + c];
        sd += v * att_dst[h * C + c];
    }
    a_src[i] = ss;
    a_dst[i] = sd;
}

// monotone float<->int key for atomicMax-based segment max
__device__ __forceinline__ int f2key(float f) {
    int i = __float_as_int(f);
    return i >= 0 ? i : (i ^ 0x7FFFFFFF);
}
__device__ __forceinline__ float key2f(int k) {
    int i = k >= 0 ? k : (k ^ 0x7FFFFFFF);
    return __int_as_float(i);
}

// pass1: raw alpha (leaky relu) + segment max per (dst, head)
__global__ void edge_pass1_kernel(const int* __restrict__ ei, const float* __restrict__ edge_attr,
                                  const float* __restrict__ scal, int koff,
                                  const float* __restrict__ a_src, const float* __restrict__ a_dst,
                                  float* __restrict__ alpha, int* __restrict__ mkey,
                                  int N, int E)
{
    long long e = (long long)blockIdx.x * blockDim.x + threadIdx.x;
    long long Epr = (long long)E + N;
    if (e >= Epr) return;
    int s, d; float ea;
    if (e < E) { s = ei[e]; d = ei[(long long)E + e]; ea = edge_attr[e]; }
    else       { s = d = (int)(e - E); ea = scal[13]; }
#pragma unroll
    for (int h = 0; h < HEADS; ++h) {
        float v = a_src[(long long)s * HEADS + h] + a_dst[(long long)d * HEADS + h]
                + ea * scal[koff + h];
        v = v > 0.0f ? v : NEG_SLOPE * v;
        alpha[e * HEADS + h] = v;
        atomicMax(&mkey[(long long)d * HEADS + h], f2key(v));
    }
}

// pass2: alpha = exp(alpha - max); segment sum per (dst, head)
__global__ void edge_pass2_kernel(const int* __restrict__ ei, float* __restrict__ alpha,
                                  const int* __restrict__ mkey, float* __restrict__ ssum,
                                  int N, int E)
{
    long long e = (long long)blockIdx.x * blockDim.x + threadIdx.x;
    long long Epr = (long long)E + N;
    if (e >= Epr) return;
    int d = (e < E) ? ei[(long long)E + e] : (int)(e - E);
#pragma unroll
    for (int h = 0; h < HEADS; ++h) {
        float m = key2f(mkey[(long long)d * HEADS + h]);
        float v = expf(alpha[e * HEADS + h] - m);
        alpha[e * HEADS + h] = v;
        atomicAdd(&ssum[(long long)d * HEADS + h], v);
    }
}

// pass3: G[dst, h, c] += P[src, h, c] * alpha/sum. One thread per (e,h,c).
// HC is a power of two (128 or 256): hcShift = log2(HC), C = HC/4.
__global__ void edge_pass3_kernel(const int* __restrict__ ei, const float* __restrict__ alpha,
                                  const float* __restrict__ ssum,
                                  const float* __restrict__ P, float* __restrict__ G,
                                  int N, int E, int hcShift)
{
    long long idx = (long long)blockIdx.x * blockDim.x + threadIdx.x;
    const int HC = 1 << hcShift;
    long long total = ((long long)E + N) << hcShift;
    if (idx >= total) return;
    long long e = idx >> hcShift;
    int rem = (int)(idx & (HC - 1));
    int h = rem >> (hcShift - 2);
    int s, d;
    if (e < E) { s = ei[e]; d = ei[(long long)E + e]; }
    else       { s = d = (int)(e - E); }
    float w = alpha[e * HEADS + h] / ssum[(long long)d * HEADS + h];
    atomicAdd(&G[(long long)d * HC + rem], P[(long long)s * HC + rem] * w);
}

// mean over heads + gat bias -> LayerNorm -> + identity -> ReLU
__global__ void gat_finish_kernel(const float* __restrict__ G, const float* __restrict__ gbias,
                                  const float* __restrict__ lnw, const float* __restrict__ lnb,
                                  const float* __restrict__ iden, float* __restrict__ out,
                                  int N, int C)
{
    int n = blockIdx.x * blockDim.x + threadIdx.x;
    if (n >= N) return;
    float t[64];
    const float* g = G + (long long)n * HEADS * C;
    float mu = 0.0f;
    for (int c = 0; c < C; ++c) {
        float v = 0.25f * (g[c] + g[C + c] + g[2 * C + c] + g[3 * C + c]) + gbias[c];
        t[c] = v; mu += v;
    }
    mu /= (float)C;
    float var = 0.0f;
    for (int c = 0; c < C; ++c) { float dd = t[c] - mu; var += dd * dd; }
    var /= (float)C;
    float r = rsqrtf(var + EPS);
    for (int c = 0; c < C; ++c) {
        float y = (t[c] - mu) * r * lnw[c] + lnb[c] + iden[(long long)n * C + c];
        out[(long long)n * C + c] = fmaxf(y, 0.0f);
    }
}

// plain LayerNorm (+ optional ReLU), in-place safe (thread owns its row)
__global__ void ln_relu_kernel(const float* __restrict__ in, const float* __restrict__ w,
                               const float* __restrict__ b, float* __restrict__ out,
                               int N, int C, int relu)
{
    int n = blockIdx.x * blockDim.x + threadIdx.x;
    if (n >= N) return;
    float t[64];
    float mu = 0.0f;
    for (int c = 0; c < C; ++c) { t[c] = in[(long long)n * C + c]; mu += t[c]; }
    mu /= (float)C;
    float var = 0.0f;
    for (int c = 0; c < C; ++c) { float dd = t[c] - mu; var += dd * dd; }
    var /= (float)C;
    float r = rsqrtf(var + EPS);
    for (int c = 0; c < C; ++c) {
        float y = (t[c] - mu) * r * w[c] + b[c];
        if (relu) y = fmaxf(y, 0.0f);
        out[(long long)n * C + c] = y;
    }
}

// final classifier: logits[n,3] = h[n,:32]@c2_w^T + c2_b, then log_softmax
__global__ void logits_kernel(const float* __restrict__ Hf, const float* __restrict__ w,
                              const float* __restrict__ b, float* __restrict__ out, int N)
{
    int n = blockIdx.x * blockDim.x + threadIdx.x;
    if (n >= N) return;
    float z[3];
#pragma unroll
    for (int j = 0; j < 3; ++j) {
        float s = b[j];
        for (int c = 0; c < 32; ++c) s += Hf[(long long)n * 32 + c] * w[j * 32 + c];
        z[j] = s;
    }
    float m = fmaxf(z[0], fmaxf(z[1], z[2]));
    float lse = m + logf(expf(z[0] - m) + expf(z[1] - m) + expf(z[2] - m));
#pragma unroll
    for (int j = 0; j < 3; ++j) out[(long long)n * 3 + j] = z[j] - lse;
}

// ---------------------------------------------------------------------------
// Host orchestration: shared (non-GEMM) part of a GAT layer
// ---------------------------------------------------------------------------
static void run_gat_edges(int C,
                          const float* att_src_p, const float* att_dst_p,
                          const float* gbias, const float* lnw, const float* lnb,
                          const float* iden, float* xout,
                          float* P, float* G, float* a_src, float* a_dst,
                          int* mkey, float* ssum, float* alpha, float* scal, int koff,
                          const int* ei, const float* edge_attr,
                          int N, int E, hipStream_t stream)
{
    const int HC = HEADS * C;
    const int hcShift = (HC == 128) ? 7 : 8;
    const long long Epr = (long long)E + N;

    long long nh = (long long)N * HEADS;
    attn_node_kernel<<<(unsigned)((nh + 255) / 256), 256, 0, stream>>>(
        P, att_src_p, att_dst_p, a_src, a_dst, N, C);

    fill_i32_kernel<<<(unsigned)((nh + 255) / 256), 256, 0, stream>>>(mkey, (int)0x80000000, nh);
    fill_f32_kernel<<<(unsigned)((nh + 255) / 256), 256, 0, stream>>>(ssum, 0.0f, nh);
    long long gtot = (long long)N * HC;
    fill_f32_kernel<<<(unsigned)((gtot + 255) / 256), 256, 0, stream>>>(G, 0.0f, gtot);

    edge_pass1_kernel<<<(unsigned)((Epr + 255) / 256), 256, 0, stream>>>(
        ei, edge_attr, scal, koff, a_src, a_dst, alpha, mkey, N, E);
    edge_pass2_kernel<<<(unsigned)((Epr + 255) / 256), 256, 0, stream>>>(
        ei, alpha, mkey, ssum, N, E);
    long long wtot = Epr << hcShift;
    edge_pass3_kernel<<<(unsigned)((wtot + 255) / 256), 256, 0, stream>>>(
        ei, alpha, ssum, P, G, N, E, hcShift);

    gat_finish_kernel<<<(unsigned)((N + 255) / 256), 256, 0, stream>>>(
        G, gbias, lnw, lnb, iden, xout, N, C);
}

extern "C" void kernel_launch(void* const* d_in, const int* in_sizes, int n_in,
                              void* d_out, int out_size, void* d_ws, size_t ws_size,
                              hipStream_t stream)
{
    (void)n_in; (void)out_size;
    const float* x         = (const float*)d_in[0];
    const float* edge_attr = (const float*)d_in[1];
    const float* in_w = (const float*)d_in[2];
    const float* in_b = (const float*)d_in[3];
    // conv1
    const float* c1_lin = (const float*)d_in[4];
    const float* c1_le  = (const float*)d_in[5];
    const float* c1_as  = (const float*)d_in[6];
    const float* c1_ad  = (const float*)d_in[7];
    const float* c1_ae  = (const float*)d_in[8];
    const float* c1_bia = (const float*)d_in[9];
    const float* n1_w = (const float*)d_in[10];
    const float* n1_b = (const float*)d_in[11];
    // conv2
    const float* c2_lin = (const float*)d_in[12];
    const float* c2_le  = (const float*)d_in[13];
    const float* c2_as  = (const float*)d_in[14];
    const float* c2_ad  = (const float*)d_in[15];
    const float* c2_ae  = (const float*)d_in[16];
    const float* c2_bia = (const float*)d_in[17];
    const float* n2_w = (const float*)d_in[18];
    const float* n2_b = (const float*)d_in[19];
    const float* p2_w = (const float*)d_in[20];
    const float* p2_b = (const float*)d_in[21];
    // conv3
    const float* c3_lin = (const float*)d_in[22];
    const float* c3_le  = (const float*)d_in[23];
    const float* c3_as  = (const float*)d_in[24];
    const float* c3_ad  = (const float*)d_in[25];
    const float* c3_ae  = (const float*)d_in[26];
    const float* c3_bia = (const float*)d_in[27];
    const float* n3_w = (const float*)d_in[28];
    const float* n3_b = (const float*)d_in[29];
    const float* p3_w = (const float*)d_in[30];
    const float* p3_b = (const float*)d_in[31];
    const float* cls1_w = (const float*)d_in[32];
    const float* cls1_b = (const float*)d_in[33];
    const float* cln_w  = (const float*)d_in[34];
    const float* cln_b  = (const float*)d_in[35];
    const float* cls2_w = (const float*)d_in[36];
    const float* cls2_b = (const float*)d_in[37];
    const int*   ei     = (const int*)d_in[38];   // edge_index [2,E] (row0 src, row1 dst)

    const int N = in_sizes[0] / 16;               // x is [N, F_IN=16]
    const int E = in_sizes[1];                    // edge_attr is [E,1]

    // ---- workspace layout (floats) ----
    size_t off = 0;
    char* base = (char*)d_ws;
    auto alloc = [&](size_t nfloats) -> float* {
        float* p = (float*)(base + off);
        off += nfloats * sizeof(float);
        return p;
    };
    float* buf_x  = alloc((size_t)N * 64);        // current activation (<=64 wide)
    float* buf_id = alloc((size_t)N * 64);        // identity / temp
    float* buf_P  = alloc((size_t)N * 256);       // GAT projection [N, H*C]
    float* buf_G  = alloc((size_t)N * 256);       // aggregation     [N, H*C]
    float* a_src  = alloc((size_t)N * HEADS);
    float* a_dst  = alloc((size_t)N * HEADS);
    int*   mkey   = (int*)alloc((size_t)N * HEADS);
    float* ssum   = alloc((size_t)N * HEADS);
    float* alpha  = alloc(((size_t)E + N) * HEADS);
    float* scal   = alloc(16);
    if (off > ws_size) return;                    // insufficient scratch: bail

    // ---- scalar precompute: edge_attr mean + per-head edge-attention k ----
    fill_f32_kernel<<<1, 32, 0, stream>>>(scal, 0.0f, 16);
    int rblocks = (E + 255) / 256; if (rblocks > 4096) rblocks = 4096;
    reduce_sum_kernel<<<(unsigned)rblocks, 256, 0, stream>>>(edge_attr, E, scal);
    finalize_scalars_kernel<<<1, 32, 0, stream>>>(c1_le, c1_ae, c2_le, c2_ae, c3_le, c3_ae,
                                                  scal, 1.0f / (float)E);

    // ---- input projection: x = relu(x @ in_w^T + in_b)  [N,16]->[N,32] ----
    launch_gemm<16, 32, true, true>(x, in_w, in_b, buf_x, N, stream);

    // ---- layer 1: GAT(32->32), identity = x (in-place safe) ----
    launch_gemm<32, 128, false, false>(buf_x, c1_lin, nullptr, buf_P, N, stream);
    run_gat_edges(32, c1_as, c1_ad, c1_bia, n1_w, n1_b,
                  buf_x, buf_x, buf_P, buf_G, a_src, a_dst, mkey, ssum, alpha, scal, 1,
                  ei, edge_attr, N, E, stream);

    // ---- layer 2: identity = x @ p2_w^T + p2_b; GAT(32->64) ----
    launch_gemm<32, 64, true, false>(buf_x, p2_w, p2_b, buf_id, N, stream);
    launch_gemm<32, 256, false, false>(buf_x, c2_lin, nullptr, buf_P, N, stream);
    run_gat_edges(64, c2_as, c2_ad, c2_bia, n2_w, n2_b,
                  buf_id, buf_x, buf_P, buf_G, a_src, a_dst, mkey, ssum, alpha, scal, 5,
                  ei, edge_attr, N, E, stream);

    // ---- layer 3: identity = x @ p3_w^T + p3_b; GAT(64->32) ----
    launch_gemm<64, 32, true, false>(buf_x, p3_w, p3_b, buf_id, N, stream);
    launch_gemm<64, 128, false, false>(buf_x, c3_lin, nullptr, buf_P, N, stream);
    run_gat_edges(32, c3_as, c3_ad, c3_bia, n3_w, n3_b,
                  buf_id, buf_x, buf_P, buf_G, a_src, a_dst, mkey, ssum, alpha, scal, 9,
                  ei, edge_attr, N, E, stream);

    // ---- classifier head ----
    launch_gemm<32, 32, true, false>(buf_x, cls1_w, cls1_b, buf_id, N, stream);
    ln_relu_kernel<<<(unsigned)((N + 255) / 256), 256, 0, stream>>>(buf_id, cln_w, cln_b,
                                                                    buf_id, N, 32, 1);
    logits_kernel<<<(unsigned)((N + 255) / 256), 256, 0, stream>>>(buf_id, cls2_w, cls2_b,
                                                                   (float*)d_out, N);
}
